// DensityMap_15616501088354
// MI455X (gfx1250) — matrix operand order repaired and checked by reading, not compiled, not run on hardware
//
#include <hip/hip_runtime.h>
#include <cmath>

typedef __attribute__((ext_vector_type(16))) _Float16 v16h;
typedef __attribute__((ext_vector_type(8)))  _Float16 v8h;
typedef __attribute__((ext_vector_type(2)))  _Float16 v2h;
typedef __attribute__((ext_vector_type(8)))  float    v8f;

#define BB   64
#define VV   2048
#define GG   256
#define TILE 128    // output tile per block (128x128)
#define KC   64     // K-chunk (cells per LDS stage)
#define LDK  72     // padded LDS row stride in halves (144B, 16B aligned)

struct GaussW { float w[13]; };

static __device__ __forceinline__ v16h cat8(v8h lo, v8h hi) {
    v16h r;
#pragma unroll
    for (int i = 0; i < 8; ++i) { r[i] = lo[i]; r[i + 8] = hi[i]; }
    return r;
}

static __device__ __forceinline__ float fast_sigmoid(float t) {
    // 1/(1+e^-t): v_exp_f32 + v_rcp_f32, no IEEE divide expansion
    return __builtin_amdgcn_rcpf(1.0f + __expf(-t));
}

// ---------------------------------------------------------------------------
// Kernel 1: fused sigmoid-rasterize + WMMA einsum  density[b,y,x] = sum_v Y*X
// grid = (4 tiles of 128x128, 64 batches), block = 256 threads (8 waves)
// Each wave: 2x4 grid of 16x16 accumulators (A-fragment reuse across 4 B's)
// ---------------------------------------------------------------------------
__global__ __launch_bounds__(256) void density_wmma_kernel(
    const float* __restrict__ pos,          // [B][V][2]
    const float* __restrict__ sz,           // [V][2]
    const unsigned char* __restrict__ mask, // [V] (bool)
    float* __restrict__ dens)               // [B][G][G] raw density
{
    __shared__ _Float16 Ysh[TILE][LDK];     // [y-in-tile][k]  (A source)
    __shared__ _Float16 Xsh[TILE][LDK];     // [x-in-tile][k]  (B source)
    __shared__ float spx[KC], spy[KC], shx[KC], shy[KC], smk[KC];

    const int b      = blockIdx.y;
    const int tileId = blockIdx.x;          // 0..3 -> 2x2 tiles of 128x128
    const int yO     = (tileId >> 1) * TILE;
    const int xO     = (tileId & 1) * TILE;
    const int t      = threadIdx.x;
    const int lane   = t & 31;
    const int wave   = t >> 5;              // 0..7
    const int wr2    = (wave & 3) * 2;      // first of two wmma tile rows
    const int wc4    = (wave >> 2) * 4;     // first of four wmma tile cols

    const int half = lane >> 4;             // lane half (0/1)
    const int mrow = lane & 15;             // M (for A) / N (for B/C)

    v8f acc[2][4] = {};

    for (int k0 = 0; k0 < VV; k0 += KC) {
        __syncthreads();                    // protect previous chunk's reads
        if (t < KC) {                       // stage per-cell params
            int v = k0 + t;
            spx[t] = (pos[((size_t)b * VV + v) * 2 + 0] + 1.0f) * 0.5f * (GG - 1);
            spy[t] = (pos[((size_t)b * VV + v) * 2 + 1] + 1.0f) * 0.5f * (GG - 1);
            shx[t] = sz[v * 2 + 0] * (GG * 0.25f);   // grid_size/2
            shy[t] = sz[v * 2 + 1] * (GG * 0.25f);
            smk[t] = mask[v] ? 1.0f : 0.0f;
        }
        __syncthreads();
        // build f16 operand tiles in LDS; two k's per thread -> packed b32 store
#pragma unroll
        for (int i = 0; i < (KC / 2) * TILE / 256; ++i) {   // 16 iters
            int idx = t + i * 256;
            int kk  = (idx / TILE) * 2;     // even cell index within chunk
            int e   = idx % TILE;           // grid coordinate within tile
            float cx = (float)(xO + e);
            v2h px;
            px[0] = (_Float16)fast_sigmoid((shx[kk]     - fabsf(cx - spx[kk]))     * 2.0f);
            px[1] = (_Float16)fast_sigmoid((shx[kk + 1] - fabsf(cx - spx[kk + 1])) * 2.0f);
            *(v2h*)&Xsh[e][kk] = px;
            float cy = (float)(yO + e);
            v2h py;
            py[0] = (_Float16)(smk[kk]     * fast_sigmoid((shy[kk]     - fabsf(cy - spy[kk]))     * 2.0f));
            py[1] = (_Float16)(smk[kk + 1] * fast_sigmoid((shy[kk + 1] - fabsf(cy - spy[kk + 1])) * 2.0f));
            *(v2h*)&Ysh[e][kk] = py;
        }
        __syncthreads();
        // two K-steps of v_wmma_f32_16x16x32_f16, 8 tiles per wave
#pragma unroll
        for (int ks = 0; ks < KC; ks += 32) {
            v16h af[2];
#pragma unroll
            for (int j = 0; j < 2; ++j) {
                // A fragment (16x32): lane half h holds K = h*8+i (i<8), 16+h*8+(i-8)
                const _Float16* ar = &Ysh[(wr2 + j) * 16 + mrow][ks + half * 8];
                af[j] = cat8(*(const v8h*)ar, *(const v8h*)(ar + 16));
            }
#pragma unroll
            for (int c = 0; c < 4; ++c) {
                // B fragment (32x16): lane half h holds K = h*8+i (i<8), 16+h*8+(i-8)
                const _Float16* br = &Xsh[(wc4 + c) * 16 + mrow][ks + half * 8];
                v16h bf = cat8(*(const v8h*)br, *(const v8h*)(br + 16));
#pragma unroll
                for (int j = 0; j < 2; ++j)
                    acc[j][c] = __builtin_amdgcn_wmma_f32_16x16x32_f16(
                        false, af[j], false, bf, (short)0, acc[j][c], false, false);
            }
        }
    }

    // C/D layout: VGPR r -> M = r + 8*half ; N = lane&15
#pragma unroll
    for (int j = 0; j < 2; ++j) {
#pragma unroll
        for (int c = 0; c < 4; ++c) {
            float* o = dens + (size_t)b * GG * GG
                            + (size_t)(yO + (wr2 + j) * 16 + half * 8) * GG
                            + xO + (wc4 + c) * 16 + mrow;
#pragma unroll
            for (int r = 0; r < 8; ++r) o[(size_t)r * GG] = acc[j][c][r];
        }
    }
}

// ---------------------------------------------------------------------------
// Kernel 2: fused separable 13-tap Gaussian blur (reflect pad) + overflow loss
// grid = (64 tiles of 32x32, 64 batches), block = 256
// ---------------------------------------------------------------------------
#define BT 32

static __device__ __forceinline__ int refl(int q) {
    q = (q < 0) ? -q : q;
    q = (q > GG - 1) ? (2 * (GG - 1) - q) : q;
    return q;
}

__global__ __launch_bounds__(256) void blur_loss_kernel(
    const float* __restrict__ dens, float* __restrict__ out,
    float* __restrict__ loss_acc, GaussW gw)
{
    __shared__ float sin_[BT + 12][BT + 12 + 1];  // 44 x 45 input+halo
    __shared__ float stmp[BT + 12][BT + 1];       // 44 x 33 h-blurred
    __shared__ float sred[256];

    const int b    = blockIdx.y;
    const int tile = blockIdx.x;                  // 8x8 tiles of 32x32
    const int y0   = (tile >> 3) * BT;
    const int x0   = (tile & 7) * BT;
    const float* src = dens + (size_t)b * GG * GG;

    for (int idx = threadIdx.x; idx < 44 * 44; idx += 256) {
        int ly = idx / 44, lx = idx % 44;
        sin_[ly][lx] = src[refl(y0 + ly - 6) * GG + refl(x0 + lx - 6)];
    }
    __syncthreads();

    for (int idx = threadIdx.x; idx < 44 * BT; idx += 256) {
        int ly = idx / BT, lx = idx % BT;
        float s = 0.0f;
#pragma unroll
        for (int i = 0; i < 13; ++i) s += gw.w[i] * sin_[ly][lx + i];
        stmp[ly][lx] = s;
    }
    __syncthreads();

    float lsum = 0.0f;
    for (int idx = threadIdx.x; idx < BT * BT; idx += 256) {
        int ly = idx / BT, lx = idx % BT;
        float s = 0.0f;
#pragma unroll
        for (int i = 0; i < 13; ++i) s += gw.w[i] * stmp[ly + i][lx];
        out[(size_t)b * GG * GG + (size_t)(y0 + ly) * GG + (x0 + lx)] = s;
        float ov = s - 1.0f;
        ov = ov > 0.0f ? ov : 0.0f;
        lsum += ov * ov;
    }

    sred[threadIdx.x] = lsum;
    __syncthreads();
    for (int s2 = 128; s2 > 0; s2 >>= 1) {
        if (threadIdx.x < s2) sred[threadIdx.x] += sred[threadIdx.x + s2];
        __syncthreads();
    }
    if (threadIdx.x == 0) atomicAdd(loss_acc, sred[0]);
}

__global__ void finalize_loss_kernel(const float* __restrict__ acc,
                                     float* __restrict__ dst) {
    *dst = *acc * (1.0f / ((float)BB * GG * GG));
}

// ---------------------------------------------------------------------------
extern "C" void kernel_launch(void* const* d_in, const int* in_sizes, int n_in,
                              void* d_out, int out_size, void* d_ws, size_t ws_size,
                              hipStream_t stream) {
    (void)in_sizes; (void)n_in; (void)out_size; (void)ws_size;
    const float* pos          = (const float*)d_in[0];   // [B][V][2]
    const float* sz           = (const float*)d_in[1];   // [V][2]
    const unsigned char* mask = (const unsigned char*)d_in[2]; // [V] bool
    float* out = (float*)d_out;                          // [B*G*G] + 1 (loss)

    float* dens_raw = (float*)d_ws;                      // 16 MB raw density
    float* loss_acc = (float*)((char*)d_ws + (size_t)BB * GG * GG * sizeof(float));

    hipMemsetAsync(loss_acc, 0, sizeof(float), stream);

    density_wmma_kernel<<<dim3(4, BB), 256, 0, stream>>>(pos, sz, mask, dens_raw);

    // normalized separable Gaussian weights (sigma=2, 13 taps)
    GaussW gw;
    float wsum = 0.0f;
    for (int i = 0; i < 13; ++i) { float x = (float)(i - 6); gw.w[i] = expf(-x * x / 8.0f); wsum += gw.w[i]; }
    for (int i = 0; i < 13; ++i) gw.w[i] /= wsum;

    blur_loss_kernel<<<dim3(64, BB), 256, 0, stream>>>(dens_raw, out, loss_acc, gw);
    finalize_loss_kernel<<<1, 1, 0, stream>>>(loss_acc, out + (size_t)BB * GG * GG);
}
